// EncoderImagePrecomp_45629732553080
// MI455X (gfx1250) — matrix operand order repaired and checked by reading, not compiled, not run on hardware
//
#include <hip/hip_runtime.h>
#include <math.h>

// ---------------------------------------------------------------------------
// CDNA5 (gfx1250) implementation: all GEMMs via v_wmma_f32_16x16x32_bf16.
// wave32; each wave computes a 64x64 output tile (4x4 WMMA frags -> 16 WMMAs
// per 16 b128 loads per K-step), 8 waves per 256-thread block arranged
// 2(M) x 4(N) -> 128(M) x 256(N) block tile.
// ---------------------------------------------------------------------------

typedef __bf16 bf16_t;
typedef __attribute__((ext_vector_type(8)))  __bf16 v8bf;
typedef __attribute__((ext_vector_type(16))) __bf16 v16bf;
typedef __attribute__((ext_vector_type(8)))  float  v8f;

#define B_   256
#define N_   36
#define IMG_ 2048
#define E_   1024
#define DK_  128
#define H3_  3072
#define BN_  (B_ * N_)   /* 9216 */

// --------------------------- fp32 -> bf16 conversion -----------------------
__global__ void cvt_f32_bf16_kernel(const float* __restrict__ src,
                                    bf16_t* __restrict__ dst, int n) {
    int i = blockIdx.x * blockDim.x + threadIdx.x;
    int stride = gridDim.x * blockDim.x;
    for (; i < n; i += stride) dst[i] = (bf16_t)src[i];
}

// --------------------------- WMMA bf16 GEMM --------------------------------
// C[M,N] = A[M,K] * B[N,K]^T + bias[N]   (A: row stride lda, B: row stride ldb)
__launch_bounds__(256)
__global__ void wmma_gemm_bf16_kernel(const bf16_t* __restrict__ A, int lda,
                                      const bf16_t* __restrict__ Bw, int ldb,
                                      const float* __restrict__ bias,
                                      float* __restrict__ C, int ldc,
                                      int M, int N, int K) {
    const int lane = threadIdx.x & 31;
    const int wave = threadIdx.x >> 5;                  // 0..7
    const int m0 = blockIdx.x * 128 + (wave & 1) * 64;  // wave tile: 64x64
    const int n0 = blockIdx.y * 256 + (wave >> 1) * 64;
    if (m0 >= M || n0 >= N) return;                     // wave-uniform exit

    const int lhalf = (lane >= 16) ? 1 : 0;
    const int l15   = lane & 15;

    v8f acc[4][4];
#pragma unroll
    for (int ti = 0; ti < 4; ++ti)
#pragma unroll
        for (int tj = 0; tj < 4; ++tj) acc[ti][tj] = (v8f){};

    const bf16_t* arow[4];
    const bf16_t* brow[4];
#pragma unroll
    for (int t = 0; t < 4; ++t) {
        arow[t] = A  + (size_t)(m0 + t * 16 + l15) * lda;
        brow[t] = Bw + (size_t)(n0 + t * 16 + l15) * ldb;
    }

    for (int kk = 0; kk < K; kk += 32) {
        const int ka = kk + lhalf * 8;    // A: lanes 16-31 hold K+8..15 / K+24..31
        const int kb = kk + lhalf * 8;    // B: lanes 16-31 hold K+8..15 / K+24..31
        v16bf a[4], b[4];
#pragma unroll
        for (int t = 0; t < 4; ++t) {
            v8bf alo = *(const v8bf*)(arow[t] + ka);
            v8bf ahi = *(const v8bf*)(arow[t] + ka + 16);
            a[t] = __builtin_shufflevector(alo, ahi, 0,1,2,3,4,5,6,7,8,9,10,11,12,13,14,15);
        }
#pragma unroll
        for (int t = 0; t < 4; ++t) {
            v8bf blo = *(const v8bf*)(brow[t] + kb);
            v8bf bhi = *(const v8bf*)(brow[t] + kb + 16);
            b[t] = __builtin_shufflevector(blo, bhi, 0,1,2,3,4,5,6,7,8,9,10,11,12,13,14,15);
        }
#pragma unroll
        for (int ti = 0; ti < 4; ++ti)
#pragma unroll
            for (int tj = 0; tj < 4; ++tj)
                acc[ti][tj] = __builtin_amdgcn_wmma_f32_16x16x32_bf16(
                    false, a[ti], false, b[tj], (short)0, acc[ti][tj], false, false);
    }

    const int roff = lhalf * 8;                 // C/D layout: lanes 16-31 -> M+8
#pragma unroll
    for (int tj = 0; tj < 4; ++tj) {
        const int c = n0 + tj * 16 + l15;
        const float bv = bias ? bias[c] : 0.0f;
#pragma unroll
        for (int ti = 0; ti < 4; ++ti) {
#pragma unroll
            for (int i = 0; i < 8; ++i) {
                const size_t r = (size_t)(m0 + ti * 16 + i + roff) * ldc;
                C[r + c] = acc[ti][tj][i] + bv;
            }
        }
    }
}

// ------------------- row l2norm + broadcast oimages add --------------------
// one block per row of f (BN_ rows, E_ cols); also writes bf16 copy
__global__ void l2norm_add_kernel(float* __restrict__ f, bf16_t* __restrict__ fb,
                                  const float* __restrict__ oimg) {
    const int row = blockIdx.x;
    const int b = row / N_;
    __shared__ float wsum[8];
    __shared__ float nrm;
    float* fr = f + (size_t)row * E_;
    float s = 0.0f;
    for (int d = threadIdx.x; d < E_; d += 256) { float v = fr[d]; s += v * v; }
    for (int off = 16; off > 0; off >>= 1) s += __shfl_down(s, off, 32);
    if ((threadIdx.x & 31) == 0) wsum[threadIdx.x >> 5] = s;
    __syncthreads();
    if (threadIdx.x == 0) {
        float t = 0.0f;
        for (int i = 0; i < 8; ++i) t += wsum[i];
        nrm = sqrtf(t) + 1e-8f;
    }
    __syncthreads();
    const float inv = 1.0f / nrm;
    const float* ob = oimg + (size_t)b * E_;
    bf16_t* fbr = fb + (size_t)row * E_;
    for (int d = threadIdx.x; d < E_; d += 256) {
        float v = fr[d] * inv + ob[d];
        fr[d] = v;
        fbr[d] = (bf16_t)v;
    }
}

// ---- attention residual: f += rowsum(softmax(k qT /sqrt(dk))) * v  --------
// one block per (b,i); scores over j=0..35 are tiny -> scalar path
__global__ void attn_kernel(const float* __restrict__ kb, const float* __restrict__ qb,
                            const float* __restrict__ vb,
                            float* __restrict__ f, bf16_t* __restrict__ fb) {
    const int bi = blockIdx.x;       // b*N_ + i
    const int b  = bi / N_;
    __shared__ float sc[N_];
    __shared__ float srow;
    const int tid = threadIdx.x;
    if (tid < N_) {
        const float* kr = kb + (size_t)bi * DK_;
        const float* qr = qb + ((size_t)b * N_ + tid) * DK_;
        float s = 0.0f;
        for (int d = 0; d < DK_; ++d) s += kr[d] * qr[d];
        sc[tid] = s * 0.08838834764831845f;   // 1/sqrt(128)
    }
    __syncthreads();
    if (tid == 0) {
        float m = sc[0];
        for (int j = 1; j < N_; ++j) m = fmaxf(m, sc[j]);
        float Z = 0.0f;
        for (int j = 0; j < N_; ++j) Z += expf(sc[j] - m);
        float r = 0.0f;
        for (int j = 0; j < N_; ++j) r += expf(sc[j] - m) / Z;  // ~1.0, faithful
        srow = r;
    }
    __syncthreads();
    const float r = srow;
    const float* vr = vb + (size_t)bi * E_;
    float* fr = f + (size_t)bi * E_;
    bf16_t* fbr = fb + (size_t)bi * E_;
    for (int d = tid; d < E_; d += 256) {
        float val = fr[d] + r * vr[d];
        fr[d] = val;
        fbr[d] = (bf16_t)val;
    }
}

// ----------------------------- GRU pieces ----------------------------------
__global__ void zero_h_kernel(float* __restrict__ h, bf16_t* __restrict__ hb, int n) {
    int i = blockIdx.x * blockDim.x + threadIdx.x;
    if (i < n) { h[i] = 0.0f; hb[i] = (bf16_t)0.0f; }
}

// torch gate order r,z,n ; gi/gh are (B_, 3E) with biases already added by GEMM
__global__ void gru_gate_kernel(const float* __restrict__ gi, const float* __restrict__ gh,
                                float* __restrict__ h, bf16_t* __restrict__ hb) {
    const int idx = blockIdx.x * blockDim.x + threadIdx.x;   // < B_*E_
    const int b = idx >> 10;
    const int d = idx & (E_ - 1);
    const float* gib = gi + (size_t)b * H3_;
    const float* ghb = gh + (size_t)b * H3_;
    const float ir = gib[d], iz = gib[d + E_], in = gib[d + 2 * E_];
    const float hr = ghb[d], hz = ghb[d + E_], hn = ghb[d + 2 * E_];
    const float r = 1.0f / (1.0f + expf(-(ir + hr)));
    const float z = 1.0f / (1.0f + expf(-(iz + hz)));
    const float n = tanhf(in + r * hn);
    const float hv = h[idx];
    const float hnew = (1.0f - z) * n + z * hv;
    h[idx] = hnew;
    hb[idx] = (bf16_t)hnew;
}

// ------------------------- final l2norm -> d_out ---------------------------
__global__ void final_l2norm_kernel(const float* __restrict__ h, float* __restrict__ out) {
    const int b = blockIdx.x;
    __shared__ float wsum[8];
    __shared__ float nrm;
    const float* hr = h + (size_t)b * E_;
    float s = 0.0f;
    for (int d = threadIdx.x; d < E_; d += 256) { float v = hr[d]; s += v * v; }
    for (int off = 16; off > 0; off >>= 1) s += __shfl_down(s, off, 32);
    if ((threadIdx.x & 31) == 0) wsum[threadIdx.x >> 5] = s;
    __syncthreads();
    if (threadIdx.x == 0) {
        float t = 0.0f;
        for (int i = 0; i < 8; ++i) t += wsum[i];
        nrm = sqrtf(t) + 1e-8f;
    }
    __syncthreads();
    const float inv = 1.0f / nrm;
    float* outr = out + (size_t)b * E_;
    for (int d = threadIdx.x; d < E_; d += 256) outr[d] = hr[d] * inv;
}

// ---------------------------------------------------------------------------
extern "C" void kernel_launch(void* const* d_in, const int* in_sizes, int n_in,
                              void* d_out, int out_size, void* d_ws, size_t ws_size,
                              hipStream_t stream) {
    (void)in_sizes; (void)n_in; (void)out_size; (void)ws_size;
    const float* images  = (const float*)d_in[0];
    const float* oimages = (const float*)d_in[1];
    const float* wl_w = (const float*)d_in[2];
    const float* wl_b = (const float*)d_in[3];
    const float* wk_w = (const float*)d_in[4];
    const float* wk_b = (const float*)d_in[5];
    const float* wq_w = (const float*)d_in[6];
    const float* wq_b = (const float*)d_in[7];
    const float* wv_w = (const float*)d_in[8];
    const float* wv_b = (const float*)d_in[9];
    const float* wih  = (const float*)d_in[10];
    const float* whh  = (const float*)d_in[11];
    const float* bih  = (const float*)d_in[12];
    const float* bhh  = (const float*)d_in[13];

    // ---- workspace carve-up (256B aligned) ----
    char* ws = (char*)d_ws;
    size_t off = 0;
    auto take = [&](size_t bytes) -> char* {
        char* p = ws + off;
        off += (bytes + 255) & ~(size_t)255;
        return p;
    };
    bf16_t* images_bf = (bf16_t*)take((size_t)BN_ * IMG_ * 2);   // 37.7 MB (later aliased by v)
    bf16_t* wl_bf  = (bf16_t*)take((size_t)E_  * IMG_ * 2);
    bf16_t* wk_bf  = (bf16_t*)take((size_t)DK_ * E_ * 2);
    bf16_t* wq_bf  = (bf16_t*)take((size_t)DK_ * E_ * 2);
    bf16_t* wv_bf  = (bf16_t*)take((size_t)E_  * E_ * 2);
    bf16_t* wih_bf = (bf16_t*)take((size_t)H3_ * E_ * 2);
    bf16_t* whh_bf = (bf16_t*)take((size_t)H3_ * E_ * 2);
    float*  f_f32  = (float*) take((size_t)BN_ * E_ * 4);
    bf16_t* f_bf   = (bf16_t*)take((size_t)BN_ * E_ * 2);
    float*  kbuf   = (float*) take((size_t)BN_ * DK_ * 4);
    float*  qbuf   = (float*) take((size_t)BN_ * DK_ * 4);
    float*  h_f32  = (float*) take((size_t)B_ * E_ * 4);
    bf16_t* h_bf   = (bf16_t*)take((size_t)B_ * E_ * 2);
    float*  gi_buf = (float*) take((size_t)B_ * H3_ * 4);
    float*  gh_buf = (float*) take((size_t)B_ * H3_ * 4);
    float*  vbuf   = (float*)images_bf;   // alias: images_bf dead after GEMM1

    auto cvt = [&](const float* s, bf16_t* d, int n) {
        int g = (n + 255) / 256; if (g > 8192) g = 8192;
        hipLaunchKernelGGL(cvt_f32_bf16_kernel, dim3(g), dim3(256), 0, stream, s, d, n);
    };
    auto gemm = [&](const bf16_t* Ap, int lda, const bf16_t* Bp, int ldb,
                    const float* bias, float* Cp, int ldc, int M, int N, int K) {
        dim3 grid((M + 127) / 128, (N + 255) / 256);
        hipLaunchKernelGGL(wmma_gemm_bf16_kernel, grid, dim3(256), 0, stream,
                           Ap, lda, Bp, ldb, bias, Cp, ldc, M, N, K);
    };

    // 1) bf16 conversions (weights + images)
    cvt(images, images_bf, BN_ * IMG_);
    cvt(wl_w, wl_bf, E_ * IMG_);
    cvt(wk_w, wk_bf, DK_ * E_);
    cvt(wq_w, wq_bf, DK_ * E_);
    cvt(wv_w, wv_bf, E_ * E_);
    cvt(wih, wih_bf, H3_ * E_);
    cvt(whh, whh_bf, H3_ * E_);

    // 2) f = images @ wl^T + wl_b ; l2norm ; += oimages
    gemm(images_bf, IMG_, wl_bf, IMG_, wl_b, f_f32, E_, BN_, E_, IMG_);
    hipLaunchKernelGGL(l2norm_add_kernel, dim3(BN_), dim3(256), 0, stream, f_f32, f_bf, oimages);

    // 3) k, q, v projections
    gemm(f_bf, E_, wk_bf, E_, wk_b, kbuf, DK_, BN_, DK_, E_);
    gemm(f_bf, E_, wq_bf, E_, wq_b, qbuf, DK_, BN_, DK_, E_);
    gemm(f_bf, E_, wv_bf, E_, wv_b, vbuf, E_, BN_, E_, E_);

    // 4) attention residual (softmax row-sum * v)
    hipLaunchKernelGGL(attn_kernel, dim3(BN_), dim3(256), 0, stream, kbuf, qbuf, vbuf, f_f32, f_bf);

    // 5) GRU scan over N_ timesteps
    hipLaunchKernelGGL(zero_h_kernel, dim3((B_ * E_ + 255) / 256), dim3(256), 0, stream,
                       h_f32, h_bf, B_ * E_);
    for (int t = 0; t < N_; ++t) {
        // gi_t = f[:, t, :] @ wih^T + bih    (strided row view of f_bf)
        gemm(f_bf + (size_t)t * E_, N_ * E_, wih_bf, E_, bih, gi_buf, H3_, B_, H3_, E_);
        // gh = h @ whh^T + bhh
        gemm(h_bf, E_, whh_bf, E_, bhh, gh_buf, H3_, B_, H3_, E_);
        hipLaunchKernelGGL(gru_gate_kernel, dim3((B_ * E_ + 255) / 256), dim3(256), 0, stream,
                           gi_buf, gh_buf, h_f32, h_bf);
    }

    // 6) final l2norm -> out (fp32, 256x1024)
    hipLaunchKernelGGL(final_l2norm_kernel, dim3(B_), dim3(256), 0, stream,
                       h_f32, (float*)d_out);
}
